// gate_5136780886429
// MI455X (gfx1250) — matrix operand order repaired and checked by reading, not compiled, not run on hardware
//
#include <hip/hip_runtime.h>
#include <hip/hip_bf16.h>
#include <cstdint>

// Problem dims (from reference)
#define B_SZ 4
#define G_SZ 4096
#define F_SZ 256
#define U_SZ 128
#define NEG_VAL (-1.0e9f)

typedef __attribute__((ext_vector_type(16))) __bf16 v16bf;
typedef __attribute__((ext_vector_type(8)))  __bf16 v8bf;
typedef __attribute__((ext_vector_type(4)))  __bf16 v4bf;
typedef __attribute__((ext_vector_type(8)))  float  v8f;

// Tiling: 128x128 output tile per block, K-step 32, 8 waves (wave32),
// each wave computes a 64(M) x 32(N) sub-tile = 4x2 fragments of 16x16.
#define TM 128
#define TN 128
#define TK 32
#define LDT 40   // padded LDS row stride in bf16 elems (80 bytes, 16B-aligned rows)

// ---- Fragment loads from LDS (bf16), per CDNA5 WMMA VGPR layouts ----
// A 16x32 bf16: lane L(0-15) row M=L holds K={0..7} in v0-3, K={16..23} in v4-7;
// lanes 16-31 same rows with K offset +8.  => two ds_load_b128 per lane.
__device__ __forceinline__ v16bf ld_fragA(const __bf16* grp, int lane) {
  int row = lane & 15;
  int k0  = (lane >> 4) << 3;              // 0 or 8
  const __bf16* p = grp + row * LDT;
  v8bf lo = *(const v8bf*)(p + k0);        // K = k0 .. k0+7
  v8bf hi = *(const v8bf*)(p + k0 + 16);   // K = k0+16 .. k0+23
  return __builtin_shufflevector(lo, hi, 0,1,2,3,4,5,6,7,8,9,10,11,12,13,14,15);
}

// B 32x16 bf16 stored N-major in LDS (Bs[n][k]): lane L(0-15) col N=L holds
// K={0..7,16..23}; lanes 16-31 hold K={8..15,24..31}.  => two ds_load_b128 per lane.
__device__ __forceinline__ v16bf ld_fragB(const __bf16* grp, int lane) {
  int col = lane & 15;
  int k0  = (lane >> 4) << 3;              // 0 or 8
  const __bf16* p = grp + col * LDT;
  v8bf lo = *(const v8bf*)(p + k0);        // K = k0 .. k0+7
  v8bf hi = *(const v8bf*)(p + k0 + 16);   // K = k0+16 .. k0+23
  return __builtin_shufflevector(lo, hi, 0,1,2,3,4,5,6,7,8,9,10,11,12,13,14,15);
}

// One K=32 step over the 128x128 block tile: 4 A-frags, 2 B-frags, 8 WMMA.
__device__ __forceinline__ void mma_step(const __bf16* As, const __bf16* Bs,
                                         v8f acc[8], int lane, int wm, int wn) {
  v16bf a[4], b[2];
#pragma unroll
  for (int mi = 0; mi < 4; ++mi)
    a[mi] = ld_fragA(As + (wm * 64 + mi * 16) * LDT, lane);
#pragma unroll
  for (int nj = 0; nj < 2; ++nj)
    b[nj] = ld_fragB(Bs + (wn * 32 + nj * 16) * LDT, lane);
#pragma unroll
  for (int mi = 0; mi < 4; ++mi)
#pragma unroll
    for (int nj = 0; nj < 2; ++nj)
      acc[mi * 2 + nj] = __builtin_amdgcn_wmma_f32_16x16x32_bf16(
          false, a[mi], false, b[nj], (short)0, acc[mi * 2 + nj], false, false);
}

// ---- Vectorized LDS tile fills (f32 global -> bf16 LDS) ----
// T[m][k] (128 x 32, stride LDT) from src[m][k] (row stride srcStride).
// Fully unrolled, b128 global loads, b64 packed DS stores.
__device__ __forceinline__ void fill_rowmajor(__bf16* __restrict__ T,
                                              const float* __restrict__ src,
                                              size_t srcStride, int tid) {
#pragma unroll
  for (int it = 0; it < (TM * TK) / (256 * 4); ++it) {
    int gi = tid + it * 256;
    int m = gi >> 3, k = (gi & 7) << 2;
    float4 v = *(const float4*)(src + (size_t)m * srcStride + k);
    v4bf p = {(__bf16)v.x, (__bf16)v.y, (__bf16)v.z, (__bf16)v.w};
    *(v4bf*)(T + m * LDT + k) = p;
  }
}

// T[n][k] (128 x 32, stride LDT) from src[k][n] (row stride srcStride).
// b128 global loads along n (coalesced), 4 scalar b16 DS scatters.
__device__ __forceinline__ void fill_transpose(__bf16* __restrict__ T,
                                               const float* __restrict__ src,
                                               size_t srcStride, int tid) {
#pragma unroll
  for (int it = 0; it < (TK * TN) / (256 * 4); ++it) {
    int gi = tid + it * 256;
    int k = gi >> 5, n = (gi & 31) << 2;
    float4 v = *(const float4*)(src + (size_t)k * srcStride + n);
    T[(n + 0) * LDT + k] = (__bf16)v.x;
    T[(n + 1) * LDT + k] = (__bf16)v.y;
    T[(n + 2) * LDT + k] = (__bf16)v.z;
    T[(n + 3) * LDT + k] = (__bf16)v.w;
  }
}

// ---------------------------------------------------------------------------
// Kernel 1: C[b, g0:+128, 0:128] = x[b] @ W   (K = F = 256). Used for f and xw2.
// ---------------------------------------------------------------------------
__global__ __launch_bounds__(256) void k_gemm_xw(const float* __restrict__ x,
                                                 const float* __restrict__ w,
                                                 float* __restrict__ out) {
  __shared__ __bf16 As[TM * LDT];
  __shared__ __bf16 Bs[TN * LDT];
  const int bb = blockIdx.y;
  const int g0 = blockIdx.x * TM;
  const float* xb = x + (size_t)bb * G_SZ * F_SZ;
  const int tid = threadIdx.x, lane = tid & 31;
  const int w_id = tid >> 5, wm = w_id >> 2, wn = w_id & 3;
  const v8f vzero = {0.f, 0.f, 0.f, 0.f, 0.f, 0.f, 0.f, 0.f};
  v8f acc[8];
#pragma unroll
  for (int i = 0; i < 8; ++i) acc[i] = vzero;

  for (int k0 = 0; k0 < F_SZ; k0 += TK) {
    __syncthreads();
    fill_rowmajor(As, xb + (size_t)g0 * F_SZ + k0, F_SZ, tid);   // A: x tile
    fill_transpose(Bs, w + (size_t)k0 * U_SZ, U_SZ, tid);        // B: w tile
    __syncthreads();
    mma_step(As, Bs, acc, lane, wm, wn);
  }

  float* ob = out + (size_t)bb * G_SZ * U_SZ;
#pragma unroll
  for (int mi = 0; mi < 4; ++mi)
#pragma unroll
    for (int nj = 0; nj < 2; ++nj) {
      v8f c = acc[mi * 2 + nj];
      int col = wn * 32 + nj * 16 + (lane & 15);
#pragma unroll
      for (int i = 0; i < 8; ++i) {
        int row = g0 + wm * 64 + mi * 16 + i + ((lane >> 4) << 3);
        ob[(size_t)row * U_SZ + col] = c[i];
      }
    }
}

// ---------------------------------------------------------------------------
// Kernel 2: gate[b,g,h] = sigmoid( f[b,g,:]·ai[:,h] + f[b,h,:]·aj[:,g]
//                                  + NEG*(1-adj[g,h]) + b_a[g,h] )
// Block computes a 128(g) x 128(h) tile: GEMM Q (M=h-rows), stash bf16 in LDS,
// GEMM P (M=g-rows) reusing staging tiles, then elementwise combine (P + Q^T).
// ---------------------------------------------------------------------------
__global__ __launch_bounds__(256) void k_gate(const float* __restrict__ f,
                                              const float* __restrict__ ai,
                                              const float* __restrict__ aj,
                                              const float* __restrict__ adj,
                                              const float* __restrict__ ba,
                                              float* __restrict__ gate) {
  __shared__ __bf16 As[TM * LDT];
  __shared__ __bf16 Bs[TN * LDT];
  __shared__ __bf16 Qt[128 * 130];      // Q tile, [local h][local g], padded
  const int bb = blockIdx.z;
  const int g0 = blockIdx.x * TM;
  const int h0 = blockIdx.y * TN;
  const float* fb = f + (size_t)bb * G_SZ * U_SZ;
  const int tid = threadIdx.x, lane = tid & 31;
  const int w_id = tid >> 5, wm = w_id >> 2, wn = w_id & 3;
  const v8f vzero = {0.f, 0.f, 0.f, 0.f, 0.f, 0.f, 0.f, 0.f};
  v8f acc[8];
#pragma unroll
  for (int i = 0; i < 8; ++i) acc[i] = vzero;

  // ---- Q = f[h0-rows] @ aj[:, g0-cols], K = U
  for (int k0 = 0; k0 < U_SZ; k0 += TK) {
    __syncthreads();
    fill_rowmajor(As, fb + (size_t)h0 * U_SZ + k0, U_SZ, tid);
    fill_transpose(Bs, aj + (size_t)k0 * G_SZ + g0, G_SZ, tid);
    __syncthreads();
    mma_step(As, Bs, acc, lane, wm, wn);
  }
  // stash Q (row = local h, col = local g)
#pragma unroll
  for (int mi = 0; mi < 4; ++mi)
#pragma unroll
    for (int nj = 0; nj < 2; ++nj) {
      v8f c = acc[mi * 2 + nj];
      int gg = wn * 32 + nj * 16 + (lane & 15);
#pragma unroll
      for (int i = 0; i < 8; ++i) {
        int hh = wm * 64 + mi * 16 + i + ((lane >> 4) << 3);
        Qt[hh * 130 + gg] = (__bf16)c[i];
      }
    }
  __syncthreads();

  // ---- P = f[g0-rows] @ ai[:, h0-cols], K = U (reuse As/Bs)
#pragma unroll
  for (int i = 0; i < 8; ++i) acc[i] = vzero;
  for (int k0 = 0; k0 < U_SZ; k0 += TK) {
    __syncthreads();
    fill_rowmajor(As, fb + (size_t)g0 * U_SZ + k0, U_SZ, tid);
    fill_transpose(Bs, ai + (size_t)k0 * G_SZ + h0, G_SZ, tid);
    __syncthreads();
    mma_step(As, Bs, acc, lane, wm, wn);
  }

  // ---- combine: attn = P + Q^T + NEG*(1-adj) + b_a ; gate = sigmoid(attn)
  float* gb = gate + (size_t)bb * G_SZ * G_SZ;
#pragma unroll
  for (int mi = 0; mi < 4; ++mi)
#pragma unroll
    for (int nj = 0; nj < 2; ++nj) {
      v8f c = acc[mi * 2 + nj];
      int hh = wn * 32 + nj * 16 + (lane & 15);
#pragma unroll
      for (int i = 0; i < 8; ++i) {
        int gg = wm * 64 + mi * 16 + i + ((lane >> 4) << 3);
        size_t gi = (size_t)(g0 + gg) * G_SZ + (h0 + hh);
        float a  = adj[gi];
        float bv = ba[gi];
        float q  = (float)Qt[hh * 130 + gg];
        float attn = c[i] + q + NEG_VAL * (1.0f - a) + bv;
        gb[gi] = 1.0f / (1.0f + __expf(-attn));
      }
    }
}

// ---------------------------------------------------------------------------
// Kernel 3: out[b,h,u] = relu( sum_g gate[b,g,h] * f[b,g,u] + xw2[b,h,u] )
// A-tile = gate^T (transpose via index-swapped LDS scatter), B-tile = f.
// ---------------------------------------------------------------------------
__global__ __launch_bounds__(256) void k_agg(const float* __restrict__ gate,
                                             const float* __restrict__ f,
                                             const float* __restrict__ xw2,
                                             float* __restrict__ out) {
  __shared__ __bf16 As[TM * LDT];
  __shared__ __bf16 Bs[TN * LDT];
  const int bb = blockIdx.y;
  const int h0 = blockIdx.x * TM;
  const float* gb = gate + (size_t)bb * G_SZ * G_SZ;
  const float* fb = f + (size_t)bb * G_SZ * U_SZ;
  const int tid = threadIdx.x, lane = tid & 31;
  const int w_id = tid >> 5, wm = w_id >> 2, wn = w_id & 3;
  const v8f vzero = {0.f, 0.f, 0.f, 0.f, 0.f, 0.f, 0.f, 0.f};
  v8f acc[8];
#pragma unroll
  for (int i = 0; i < 8; ++i) acc[i] = vzero;

  for (int k0 = 0; k0 < G_SZ; k0 += TK) {     // K runs over g
    __syncthreads();
    // A-tile is gate^T: As[h][k] = gate[g0+k][h0+h]  (transpose fill)
    fill_transpose(As, gb + (size_t)k0 * G_SZ + h0, G_SZ, tid);
    // B-tile: Bs[n][k] = f[g0+k][n]
    fill_transpose(Bs, fb + (size_t)k0 * U_SZ, U_SZ, tid);
    __syncthreads();
    mma_step(As, Bs, acc, lane, wm, wn);
  }

  const float* xb = xw2 + (size_t)bb * G_SZ * U_SZ;
  float* ob = out + (size_t)bb * G_SZ * U_SZ;
#pragma unroll
  for (int mi = 0; mi < 4; ++mi)
#pragma unroll
    for (int nj = 0; nj < 2; ++nj) {
      v8f c = acc[mi * 2 + nj];
      int col = wn * 32 + nj * 16 + (lane & 15);
#pragma unroll
      for (int i = 0; i < 8; ++i) {
        int row = h0 + wm * 64 + mi * 16 + i + ((lane >> 4) << 3);
        float v = c[i] + xb[(size_t)row * U_SZ + col];
        ob[(size_t)row * U_SZ + col] = v > 0.f ? v : 0.f;
      }
    }
}

// ---------------------------------------------------------------------------
extern "C" void kernel_launch(void* const* d_in, const int* in_sizes, int n_in,
                              void* d_out, int out_size, void* d_ws, size_t ws_size,
                              hipStream_t stream) {
  (void)in_sizes; (void)n_in; (void)out_size; (void)ws_size;
  const float* x   = (const float*)d_in[0];   // [B,G,F]
  const float* adj = (const float*)d_in[1];   // [G,G]
  const float* w   = (const float*)d_in[2];   // [F,U]
  const float* ai  = (const float*)d_in[3];   // [U,G]
  const float* aj  = (const float*)d_in[4];   // [U,G]
  const float* w2  = (const float*)d_in[5];   // [F,U]
  const float* ba  = (const float*)d_in[6];   // [G,G]

  float* out_p  = (float*)d_out;                          // [B,G,U]
  float* gate_p = out_p + (size_t)B_SZ * G_SZ * U_SZ;     // [B,G,G]

  float* fbuf = (float*)d_ws;                             // [B,G,U] f32 (8 MB)
  float* xw2  = fbuf + (size_t)B_SZ * G_SZ * U_SZ;        // [B,G,U] f32 (8 MB)

  dim3 blk(256);
  dim3 g1(G_SZ / TM, B_SZ);
  k_gemm_xw<<<g1, blk, 0, stream>>>(x, w,  fbuf);
  k_gemm_xw<<<g1, blk, 0, stream>>>(x, w2, xw2);

  dim3 g2(G_SZ / TM, G_SZ / TN, B_SZ);
  k_gate<<<g2, blk, 0, stream>>>(fbuf, ai, aj, adj, ba, gate_p);

  dim3 g3(G_SZ / TM, B_SZ);
  k_agg<<<g3, blk, 0, stream>>>(gate_p, fbuf, xw2, out_p);
}
